// FlowAssembly_28372553957669
// MI455X (gfx1250) — matrix operand order repaired and checked by reading, not compiled, not run on hardware
//
#include <hip/hip_runtime.h>
#include <hip/hip_bf16.h>

typedef __attribute__((ext_vector_type(16))) _Float16 v16h;
typedef __attribute__((ext_vector_type(8)))  float    v8f;

#define B_  4
#define N_  16384
#define C_  32
#define K_  16
#define H_  64
#define C1_ 16
#define C2_ 16
#define SC_ (2.0f * 1.9f / 3.14159265358979323846f)

__device__ __forceinline__ v8f wmma16(v16h a, v16h b, v8f c) {
  // v_wmma_f32_16x16x32_f16 : D = A(16x32 f16) * B(32x16 f16) + C(16x16 f32)
  return __builtin_amdgcn_wmma_f32_16x16x32_f16(false, a, false, b, (short)0, c,
                                                false, false);
}

__device__ __forceinline__ v8f vzero8() {
  v8f z = {0.f, 0.f, 0.f, 0.f, 0.f, 0.f, 0.f, 0.f};
  return z;
}

// ---- WMMA 16-bit A-tile register layout helpers -----------------------------
// ISA layout, tile stored lane-major in LDS: half-slot (lane, i) with
//   k = (i<8) ? (i&7) + 8*(lane>>4) : 16 + (i&7) + 8*(lane>>4),   m = lane&15
// i.e. element (m,k) lives at lane = (m&15) + 16*((k>>3)&1),
//                     slot i = ((k&16)?8:0) + (k&7).
__device__ __forceinline__ void store_A_elem(_Float16* tile, int m, int k, float v) {
  int gk = (k >> 3) & 1;
  int slot = ((k & 16) ? 8 : 0) + (k & 7);
  tile[((m & 15) + 16 * gk) * 16 + slot] = (_Float16)v;
}

__device__ __forceinline__ v16h load_A(const _Float16* tile) {
  int lane = threadIdx.x & 31;
  return *(const v16h*)(tile + lane * 16);   // lane-contiguous 32B
}

// Pack an A register directly from an f32 LDS matrix (used once per tile for
// the raw-input layer only).
__device__ __forceinline__ v16h pack_A(const float* src, int lda, int kmax) {
  int lane = threadIdx.x & 31;
  int m = lane & 15, g = lane >> 4;
  v16h a;
#pragma unroll
  for (int i = 0; i < 16; ++i) {
    int v = i >> 1, h = i & 1;
    int k = (v < 4) ? (2 * v + 8 * g + h) : (16 + 2 * (v - 4) + 8 * g + h);
    float f = (k < kmax) ? src[m * lda + k] : 0.0f;
    a[i] = (_Float16)f;
  }
  return a;
}

// Pre-pack a 32x16 B tile (row-major f32 weight W[K][ldn]) into LDS in the
// B-register layout: slot (lane,i) = element (k = (i<8) ? 8*(lane>>4)+i : 16+8*(lane>>4)+(i-8), n = lane&15).
__device__ __forceinline__ void pack_B_tile(_Float16* dst, const float* W, int ldn,
                                            int koff, int noff, int kmax) {
  for (int s = (int)threadIdx.x; s < 512; s += (int)blockDim.x) {
    int lane = s >> 4, i = s & 15;
    int g = lane >> 4, n = lane & 15;
    int k = (i < 8) ? (8 * g + i) : (16 + 8 * g + (i - 8));
    float f = (k < kmax) ? W[(koff + k) * ldn + noff + n] : 0.0f;
    dst[s] = (_Float16)f;
  }
}

__device__ __forceinline__ v16h load_B(const _Float16* tile) {
  int lane = threadIdx.x & 31;
  return *(const v16h*)(tile + lane * 16);
}

__device__ __forceinline__ float wave_sum(float v) {
#pragma unroll
  for (int off = 16; off > 0; off >>= 1) v += __shfl_xor(v, off, 32);
  return v;
}

// ---------------------------------------------------------------------------
// Kernel 1: flip + actnorm + coupling-1 MLP (16->64->64->32), writes full out:
//   out[ch 0:16]  = reverse(updated x2)   (stage-2's gather source)
//   out[ch 16:32] = reverse(x1)           (stage-2 updates this in place)
// ---------------------------------------------------------------------------
__global__ __launch_bounds__(256) void stage1_kernel(
    const float* __restrict__ x, const float* __restrict__ an_logs,
    const float* __restrict__ an_bias,
    const float* __restrict__ w1, const float* __restrict__ b1,
    const float* __restrict__ g1, const float* __restrict__ be1,
    const float* __restrict__ w2, const float* __restrict__ b2,
    const float* __restrict__ w3, const float* __restrict__ b3,
    float* __restrict__ out, float* __restrict__ ws)
{
  __shared__ __align__(32) _Float16 wt[16 * 512];   // 16 pre-packed B tiles (16KB)
  __shared__ float escale[C_];
  __shared__ float ybuf[8][16 * 32];                // per-wave actnormed tile (f32)
  __shared__ __align__(32) _Float16 habuf[8][2 * 512]; // per-wave hidden, A layout

  for (int nt = 0; nt < 4; ++nt)                    // w1: 16x64, K padded to 32
    pack_B_tile(wt + nt * 512, w1, H_, 0, nt * 16, C1_);
  for (int kc = 0; kc < 2; ++kc)                    // w2: 64x64
    for (int nt = 0; nt < 4; ++nt)
      pack_B_tile(wt + (4 + kc * 4 + nt) * 512, w2, H_, kc * 32, nt * 16, H_);
  pack_B_tile(wt + 12 * 512, w3, 32,  0,  0, H_);   // w3 s, k-chunk0
  pack_B_tile(wt + 13 * 512, w3, 32,  0, 16, H_);   // w3 t, k-chunk0
  pack_B_tile(wt + 14 * 512, w3, 32, 32,  0, H_);   // w3 s, k-chunk1
  pack_B_tile(wt + 15 * 512, w3, 32, 32, 16, H_);   // w3 t, k-chunk1
  for (int c = (int)threadIdx.x; c < C_; c += (int)blockDim.x)
    escale[c] = __expf(an_logs[c]);
  __syncthreads();

  const int wave = threadIdx.x >> 5;
  const int lane = threadIdx.x & 31;
  const int lane15 = lane & 15;
  const int g = lane >> 4;
  float* y = ybuf[wave];
  _Float16* ha = habuf[wave];
  float ldsum = 0.0f;
  const int b = blockIdx.x >> 5;                    // 32 blocks per batch

  for (int iter = 0; iter < 4; ++iter) {
    const int tile = (int)blockIdx.x * 32 + iter * 8 + wave;  // 4096 tiles total
    const long p0 = (long)tile * 16;                // global point row (b*N + n)

    // Load 16 points, flip channels, actnorm: y[c] = x[31-c]*e[c]+bias[c]
    for (int s = lane; s < 512; s += 32) {
      int m = s >> 5, c = s & 31;
      y[m * 32 + c] = x[(p0 + m) * C_ + (31 - c)] * escale[c] + an_bias[c];
    }
    __builtin_prefetch(&x[(p0 + 128) * C_], 0, 1);
    __syncthreads();

    // Layer 1: x1(16x16, K-pad 32) @ w1 ; h = relu(g1*(h+b1)+be1) -> ha (A layout)
    v16h a1 = pack_A(y, 32, C1_);
    for (int nt = 0; nt < 4; ++nt) {
      v8f d = wmma16(a1, load_B(wt + nt * 512), vzero8());
      int n = nt * 16 + lane15;
      float bb = b1[n], gg = g1[n], ee = be1[n];
      _Float16* dst = ha + (n >> 5) * 512;
      int kk = n & 31;
#pragma unroll
      for (int r = 0; r < 8; ++r) {
        float v = gg * (d[r] + bb) + ee;
        store_A_elem(dst, r + 8 * g, kk, v > 0.f ? v : 0.f);
      }
    }
    __syncthreads();

    // Layer 2: h(16x64) @ w2(64x64) + b2, relu  (A operands direct from LDS)
    {
      v16h a2a = load_A(ha);
      v16h a2b = load_A(ha + 512);
      __syncthreads();
      for (int nt = 0; nt < 4; ++nt) {
        v8f d = wmma16(a2a, load_B(wt + (4 + nt) * 512), vzero8());
        d = wmma16(a2b, load_B(wt + (8 + nt) * 512), d);
        int n = nt * 16 + lane15;
        float bb = b2[n];
        _Float16* dst = ha + (n >> 5) * 512;
        int kk = n & 31;
#pragma unroll
        for (int r = 0; r < 8; ++r) {
          float v = d[r] + bb;
          store_A_elem(dst, r + 8 * g, kk, v > 0.f ? v : 0.f);
        }
      }
      __syncthreads();
    }

    // Layer 3: h2(16x64) @ w3(64x32) -> s,t ; soft-clamp ; update x2 ; emit out
    {
      v16h a3a = load_A(ha);
      v16h a3b = load_A(ha + 512);
      v8f dsv = wmma16(a3a, load_B(wt + 12 * 512), vzero8());
      dsv = wmma16(a3b, load_B(wt + 14 * 512), dsv);
      v8f dtv = wmma16(a3a, load_B(wt + 13 * 512), vzero8());
      dtv = wmma16(a3b, load_B(wt + 15 * 512), dtv);
      float bs = b3[lane15], bt = b3[16 + lane15];
#pragma unroll
      for (int r = 0; r < 8; ++r) {
        int m = r + 8 * g;
        long p = p0 + m;
        float sv = SC_ * atanf((dsv[r] + bs) * (1.0f / SC_));
        float tv = dtv[r] + bt;
        ldsum += sv;
        float x2n = y[m * 32 + 16 + lane15] * __expf(sv) + tv;
        out[p * C_ + (15 - lane15)] = x2n;                        // reversed x2'
        out[p * C_ + 16 + lane15]   = y[m * 32 + (15 - lane15)];  // reversed x1
      }
    }
    __syncthreads();
  }

  float tot = wave_sum(ldsum);
  if (lane == 0) atomicAdd(&ws[b], tot);
}

// ---------------------------------------------------------------------------
// Kernel 2: KNN edge conv coupling. Gathers out[ch 0:16] across the batch,
// builds edge(16 nbrs x 32) directly in WMMA A layout, runs 32->64->64 (WMMA),
// max-pools over K into an A-layout buffer, batches 16 points for the 64->32
// head, updates out[ch 16:32] in place.
// ---------------------------------------------------------------------------
__global__ __launch_bounds__(256) void stage2_kernel(
    const int* __restrict__ knn,
    const float* __restrict__ k1w, const float* __restrict__ kb1,
    const float* __restrict__ k2w, const float* __restrict__ kb2,
    const float* __restrict__ k3w, const float* __restrict__ kb3,
    float* __restrict__ out, float* __restrict__ ws)
{
  __shared__ __align__(32) _Float16 wt[16 * 512];
  __shared__ __align__(32) _Float16 eabuf[8][512];      // edge tile, A layout
  __shared__ __align__(32) _Float16 habuf[8][2 * 512];  // hidden, A layout
  __shared__ __align__(32) _Float16 hrbuf[8][2 * 512];  // pooled 16x64, A layout
  __shared__ float creg[8][16];
  __shared__ int   kidx[8][16];

  for (int nt = 0; nt < 4; ++nt)                     // k1: 32x64 (exact K fit)
    pack_B_tile(wt + nt * 512, k1w, H_, 0, nt * 16, 32);
  for (int kc = 0; kc < 2; ++kc)                     // k2: 64x64
    for (int nt = 0; nt < 4; ++nt)
      pack_B_tile(wt + (4 + kc * 4 + nt) * 512, k2w, H_, kc * 32, nt * 16, H_);
  pack_B_tile(wt + 12 * 512, k3w, 32,  0,  0, H_);
  pack_B_tile(wt + 13 * 512, k3w, 32,  0, 16, H_);
  pack_B_tile(wt + 14 * 512, k3w, 32, 32,  0, H_);
  pack_B_tile(wt + 15 * 512, k3w, 32, 32, 16, H_);
  __syncthreads();

  const int wave = threadIdx.x >> 5;
  const int lane = threadIdx.x & 31;
  const int lane15 = lane & 15;
  const int g = lane >> 4;
  const int b = blockIdx.x >> 6;                     // 64 blocks per batch
  const long bbase = (long)b * N_;
  _Float16* ea = eabuf[wave];
  _Float16* ha = habuf[wave];
  _Float16* hr = hrbuf[wave];
  float ldsum = 0.0f;

  for (int iter = 0; iter < 2; ++iter) {
    const int tile = (int)blockIdx.x * 16 + iter * 8 + wave;  // 4096 tiles
    const long p0 = (long)tile * 16;

    for (int pt = 0; pt < 16; ++pt) {
      const long p = p0 + pt;
      if (lane < 16) {
        creg[wave][lane] = out[p * C_ + lane];       // center features (ch 0:16)
        kidx[wave][lane] = knn[p * K_ + lane];       // neighbor indices (in-batch)
      }
      __syncthreads();

      // edge[m][c] = c<16 ? center[c] : nb[m][c-16] - center[c-16], in A layout
      for (int s = lane; s < 512; s += 32) {
        int m = s >> 5, c = s & 31;
        float cv = creg[wave][c & 15];
        float v;
        if (c < 16) v = cv;
        else {
          long q = bbase + (long)kidx[wave][m];
          v = out[q * C_ + (c - 16)] - cv;
        }
        store_A_elem(ea, m, c, v);
      }
      __syncthreads();

      // k1: edge(16x32) @ k1(32x64) + kb1, relu -> ha (A layout)
      v16h a = load_A(ea);
      for (int nt = 0; nt < 4; ++nt) {
        v8f d = wmma16(a, load_B(wt + nt * 512), vzero8());
        int n = nt * 16 + lane15;
        float bb = kb1[n];
        _Float16* dst = ha + (n >> 5) * 512;
        int kk = n & 31;
#pragma unroll
        for (int r = 0; r < 8; ++r) {
          float v = d[r] + bb;
          store_A_elem(dst, r + 8 * g, kk, v > 0.f ? v : 0.f);
        }
      }
      __syncthreads();

      // k2 + relu, fused max over the 16 neighbor rows; pooled -> hr (A layout)
      v16h a2a = load_A(ha);
      v16h a2b = load_A(ha + 512);
      for (int nt = 0; nt < 4; ++nt) {
        v8f d = wmma16(a2a, load_B(wt + (4 + nt) * 512), vzero8());
        d = wmma16(a2b, load_B(wt + (8 + nt) * 512), d);
        int n = nt * 16 + lane15;
        float bb = kb2[n];
        float pm = 0.0f;                              // relu output lower bound
#pragma unroll
        for (int r = 0; r < 8; ++r) {
          float v = d[r] + bb;
          v = v > 0.f ? v : 0.f;
          pm = fmaxf(pm, v);
        }
        pm = fmaxf(pm, __shfl_xor(pm, 16, 32));       // merge row halves
        if (lane < 16)
          store_A_elem(hr + (n >> 5) * 512, pt, n & 31, pm);
      }
      __syncthreads();
    }

    // k3 head batched over the 16 points: hr(16x64) @ k3(64x32) -> s,t
    v16h a3a = load_A(hr);
    v16h a3b = load_A(hr + 512);
    v8f dsv = wmma16(a3a, load_B(wt + 12 * 512), vzero8());
    dsv = wmma16(a3b, load_B(wt + 14 * 512), dsv);
    v8f dtv = wmma16(a3a, load_B(wt + 13 * 512), vzero8());
    dtv = wmma16(a3b, load_B(wt + 15 * 512), dtv);
    float bs = kb3[lane15], bt = kb3[16 + lane15];
#pragma unroll
    for (int r = 0; r < 8; ++r) {
      int m = r + 8 * g;
      long p = p0 + m;
      float sv = SC_ * atanf((dsv[r] + bs) * (1.0f / SC_));
      float tv = dtv[r] + bt;
      ldsum += sv;
      float* op = &out[p * C_ + 16 + lane15];
      *op = *op * __expf(sv) + tv;
    }
    __syncthreads();
  }

  float tot = wave_sum(ldsum);
  if (lane == 0) atomicAdd(&ws[4 + b], tot);
}

__global__ void zero_ws(float* ws) {
  if (threadIdx.x < 8) ws[threadIdx.x] = 0.0f;
}

__global__ void logdet_kernel(const float* __restrict__ an_logs,
                              const float* __restrict__ ws,
                              float* __restrict__ out) {
  int lane = (int)threadIdx.x;
  float v = (lane < C_) ? an_logs[lane] : 0.0f;
  v = wave_sum(v);
  float ld0 = (float)N_ * v;
  if (lane < B_)
    out[(long)B_ * N_ * C_ + lane] = ld0 + ws[lane] + ws[4 + lane];
}

extern "C" void kernel_launch(void* const* d_in, const int* in_sizes, int n_in,
                              void* d_out, int out_size, void* d_ws, size_t ws_size,
                              hipStream_t stream) {
  const float* x       = (const float*)d_in[0];
  const int*   knn     = (const int*)  d_in[1];
  const float* an_logs = (const float*)d_in[2];
  const float* an_bias = (const float*)d_in[3];
  const float* w1  = (const float*)d_in[4];
  const float* b1  = (const float*)d_in[5];
  const float* g1  = (const float*)d_in[6];
  const float* be1 = (const float*)d_in[7];
  const float* w2  = (const float*)d_in[8];
  const float* b2  = (const float*)d_in[9];
  const float* w3  = (const float*)d_in[10];
  const float* b3  = (const float*)d_in[11];
  const float* k1  = (const float*)d_in[12];
  const float* kb1 = (const float*)d_in[13];
  const float* k2  = (const float*)d_in[14];
  const float* kb2 = (const float*)d_in[15];
  const float* k3  = (const float*)d_in[16];
  const float* kb3 = (const float*)d_in[17];
  float* out = (float*)d_out;
  float* ws  = (float*)d_ws;

  hipLaunchKernelGGL(zero_ws, dim3(1), dim3(32), 0, stream, ws);
  hipLaunchKernelGGL(stage1_kernel, dim3(128), dim3(256), 0, stream,
                     x, an_logs, an_bias, w1, b1, g1, be1, w2, b2, w3, b3, out, ws);
  hipLaunchKernelGGL(stage2_kernel, dim3(256), dim3(256), 0, stream,
                     knn, k1, kb1, k2, kb2, k3, kb3, out, ws);
  hipLaunchKernelGGL(logdet_kernel, dim3(1), dim3(32), 0, stream, an_logs, ws, out);
}
